// MultiQueryAttention_52475910422558
// MI455X (gfx1250) — matrix operand order repaired and checked by reading, not compiled, not run on hardware
//
#include <hip/hip_runtime.h>
#include <hip/hip_bf16.h>

#define N_ 2
#define L_ 2048
#define E_ 1024
#define H_ 16
#define D_ 64
#define QSCALE 0.03125f      // 1/sqrt(EMBED) = 1/32, folded into q projection
#define NEG_INF (-3.0e38f)

typedef __attribute__((ext_vector_type(16))) __bf16 v16bf;
typedef __attribute__((ext_vector_type(8)))  __bf16 v8bf;
typedef __attribute__((ext_vector_type(8)))  float  v8f;
typedef __attribute__((ext_vector_type(4)))  float  v4f;

static __device__ inline v8f wmma_bf16(v16bf a, v16bf b, v8f c) {
  // D = A(16x32 bf16) * B(32x16 bf16) + C(16x16 f32)
  return __builtin_amdgcn_wmma_f32_16x16x32_bf16(false, a, false, b, (short)0, c,
                                                 false, false);
}

// A operand (16x32 bf16) k order used here, contiguous: lane m (0..15)/m+16; lane-group g=(lane>>4):
// v16 idx 0..7  <- K = g*16 .. g*16+7
// v16 idx 8..15 <- K = g*16+8 .. g*16+15
static __device__ inline v16bf ld_a16(const __bf16* __restrict__ row, int g) {
  v8bf c0 = *(const v8bf*)(row + g * 16);
  v8bf c1 = *(const v8bf*)(row + g * 16 + 8);
  return __builtin_shufflevector(c0, c1, 0, 1, 2, 3, 4, 5, 6, 7,
                                 8, 9, 10, 11, 12, 13, 14, 15);
}

// B operand (32x16 bf16) k order used here, matching ld_a16: lane = n + 16*(k>>4), idx = k&15.
// For a fixed lane this is 16 *contiguous* elements along the contraction dim.
static __device__ inline v16bf ld_b16(const __bf16* __restrict__ p) {
  v8bf c0 = *(const v8bf*)(p);
  v8bf c1 = *(const v8bf*)(p + 8);
  return __builtin_shufflevector(c0, c1, 0, 1, 2, 3, 4, 5, 6, 7,
                                 8, 9, 10, 11, 12, 13, 14, 15);
}

static __device__ inline v8f vzero8() {
  v8f z = {0.f, 0.f, 0.f, 0.f, 0.f, 0.f, 0.f, 0.f};
  return z;
}

// ---------------------------------------------------------------------------
// W_out f32 -> bf16
__global__ void mqa_convw(const float* __restrict__ w, __bf16* __restrict__ wb, int n) {
  int i = blockIdx.x * blockDim.x + threadIdx.x;
  if (i < n) wb[i] = (__bf16)w[i];
}

// ---------------------------------------------------------------------------
// K/V projections: one block per (n,l) row; thread t computes head-dim d=t.
// kb[n][l][d] (row-major, d contiguous)  -> B matrix for QK^T
// vbt[n][d][l] (transposed, l contiguous) -> B matrix for attn@V
__global__ __launch_bounds__(64) void mqa_proj_kv(
    const float* __restrict__ keys, const float* __restrict__ values,
    const float* __restrict__ Wk, const float* __restrict__ Wv,
    __bf16* __restrict__ kb, __bf16* __restrict__ vbt) {
  __shared__ float kr[E_];
  __shared__ float vr[E_];
  int bid = blockIdx.x;           // n*L + l
  int t = threadIdx.x;            // d
  const float* krow = keys + (size_t)bid * E_;
  const float* vrow = values + (size_t)bid * E_;
  for (int j = t; j < E_; j += 64) { kr[j] = krow[j]; vr[j] = vrow[j]; }
  __syncthreads();
  const float* wkr = Wk + (size_t)t * E_;
  const float* wvr = Wv + (size_t)t * E_;
  float dk = 0.f, dv = 0.f;
  for (int f = 0; f < E_; ++f) { dk += kr[f] * wkr[f]; dv += vr[f] * wvr[f]; }
  kb[(size_t)bid * D_ + t] = (__bf16)dk;
  int n = bid >> 11;              // L_ = 2048
  int l = bid & (L_ - 1);
  vbt[((size_t)n * D_ + t) * L_ + l] = (__bf16)dv;
}

// ---------------------------------------------------------------------------
// Per-head Q projection, with 1/32 softmax scale folded in.
// qb[n][l][h][e] (e contiguous) -> A matrix rows for QK^T
__global__ __launch_bounds__(1024) void mqa_proj_q(
    const float* __restrict__ queries, const float* __restrict__ Wq,
    __bf16* __restrict__ qb) {
  __shared__ float qr[E_];
  int bid = blockIdx.x;           // n*L + l
  int t = threadIdx.x;            // h*64 + e
  qr[t] = queries[(size_t)bid * E_ + t];
  __syncthreads();
  int h = t >> 6, e = t & 63;
  const float* w = Wq + (size_t)e * D_;
  const float* q = qr + h * D_;
  float d = 0.f;
  #pragma unroll 8
  for (int i = 0; i < D_; ++i) d += q[i] * w[i];
  qb[(size_t)bid * E_ + t] = (__bf16)(d * QSCALE);
}

// ---------------------------------------------------------------------------
// Phase 1: softmax row statistics. One wave per (n, h, 16-row q-tile).
// Online (max, sumexp) per lane, cross-lane combine at the end.
__global__ __launch_bounds__(32) void mqa_stats(
    const __bf16* __restrict__ qb, const __bf16* __restrict__ kb,
    float* __restrict__ rowmax, float* __restrict__ rowsum) {
  int bid = blockIdx.x;
  int qt = bid & (L_ / 16 - 1);
  int h = (bid >> 7) & (H_ - 1);
  int n = bid >> 11;
  int lane = threadIdx.x;
  int lh = lane & 15, hf = lane >> 4;
  int q0 = qt * 16;

  const __bf16* qrow = qb + (((size_t)(n * L_ + q0 + lh)) * H_ + h) * D_;
  v16bf a0 = ld_a16(qrow, hf);          // d = 0..31
  v16bf a1 = ld_a16(qrow + 32, hf);     // d = 32..63
  const __bf16* kbase = kb + (size_t)n * L_ * D_;

  float m[8], s[8];
  #pragma unroll
  for (int r = 0; r < 8; ++r) { m[r] = NEG_INF; s[r] = 0.f; }

  for (int kc = 0; kc < L_; kc += 64) {
    v8f acc[4];
    #pragma unroll
    for (int nt = 0; nt < 4; ++nt) {
      const __bf16* krow = kbase + (size_t)(kc + nt * 16 + lh) * D_ + hf * 16;
      v16bf blo = ld_b16(krow);         // d = 0..31 half
      v16bf bhi = ld_b16(krow + 32);    // d = 32..63 half
      v8f c = vzero8();
      c = wmma_bf16(a0, blo, c);
      c = wmma_bf16(a1, bhi, c);
      acc[nt] = c;
    }
    #pragma unroll
    for (int r = 0; r < 8; ++r) {
      float x0 = acc[0][r], x1 = acc[1][r], x2 = acc[2][r], x3 = acc[3][r];
      float cm = fmaxf(fmaxf(x0, x1), fmaxf(x2, x3));
      float mn = fmaxf(m[r], cm);
      s[r] = s[r] * __expf(m[r] - mn) + __expf(x0 - mn) + __expf(x1 - mn) +
             __expf(x2 - mn) + __expf(x3 - mn);
      m[r] = mn;
    }
  }
  // combine across the 16 lanes of each half (masks 1,2,4,8 stay in-half)
  #pragma unroll
  for (int r = 0; r < 8; ++r) {
    float mv = m[r], sv = s[r];
    #pragma unroll
    for (int msk = 1; msk < 16; msk <<= 1) {
      float om = __shfl_xor(mv, msk, 32);
      float os = __shfl_xor(sv, msk, 32);
      float mm = fmaxf(mv, om);
      sv = sv * __expf(mv - mm) + os * __expf(om - mm);
      mv = mm;
    }
    if (lh == 0) {
      size_t idx = ((size_t)(n * H_ + h)) * L_ + q0 + r + 8 * hf;
      rowmax[idx] = mv;
      rowsum[idx] = sv;
    }
  }
}

// ---------------------------------------------------------------------------
// Phase 2: recompute scores, normalize, write attention ONCE (coalesced via
// LDS), and run attn@V through WMMA using the LDS tile as the A matrix.
__global__ __launch_bounds__(32) void mqa_attn(
    const __bf16* __restrict__ qb, const __bf16* __restrict__ kb,
    const __bf16* __restrict__ vbt, const float* __restrict__ rowmax,
    const float* __restrict__ rowsum, float* __restrict__ att,
    __bf16* __restrict__ oh) {
  __shared__ __align__(16) float pt[16 * 64];   // P tile: 16 q-rows x 64 k-cols
  int bid = blockIdx.x;
  int qt = bid & (L_ / 16 - 1);
  int h = (bid >> 7) & (H_ - 1);
  int n = bid >> 11;
  int lane = threadIdx.x;
  int lh = lane & 15, hf = lane >> 4;
  int q0 = qt * 16;

  const __bf16* qrow = qb + (((size_t)(n * L_ + q0 + lh)) * H_ + h) * D_;
  v16bf a0 = ld_a16(qrow, hf);
  v16bf a1 = ld_a16(qrow + 32, hf);
  const __bf16* kbase = kb + (size_t)n * L_ * D_;
  const __bf16* vb = vbt + (size_t)n * D_ * L_;
  float* arow = att + (((size_t)(n * H_ + h)) * L_ + q0) * L_;

  float m[8], rinv[8];
  #pragma unroll
  for (int r = 0; r < 8; ++r) {
    size_t idx = ((size_t)(n * H_ + h)) * L_ + q0 + r + 8 * hf;
    m[r] = rowmax[idx];
    rinv[r] = 1.f / rowsum[idx];
  }

  v8f oacc[4];
  #pragma unroll
  for (int i = 0; i < 4; ++i) oacc[i] = vzero8();

  for (int kc = 0; kc < L_; kc += 64) {
    // recompute 16x64 score tile
    v8f acc[4];
    #pragma unroll
    for (int nt = 0; nt < 4; ++nt) {
      const __bf16* krow = kbase + (size_t)(kc + nt * 16 + lh) * D_ + hf * 16;
      v16bf blo = ld_b16(krow);
      v16bf bhi = ld_b16(krow + 32);
      v8f c = vzero8();
      c = wmma_bf16(a0, blo, c);
      c = wmma_bf16(a1, bhi, c);
      acc[nt] = c;
    }
    // normalize -> probabilities, stage in LDS
    #pragma unroll
    for (int nt = 0; nt < 4; ++nt) {
      #pragma unroll
      for (int r = 0; r < 8; ++r) {
        float p = __expf(acc[nt][r] - m[r]) * rinv[r];
        pt[(r + 8 * hf) * 64 + nt * 16 + lh] = p;
      }
    }
    __syncthreads();
    // coalesced b128 stores of attention probs (written exactly once)
    {
      int row = lane >> 1, ch = lane & 1;
      const v4f* lp = (const v4f*)pt;
      v4f* gp = (v4f*)(arow + (size_t)row * L_ + kc + ch * 32);
      #pragma unroll
      for (int j = 0; j < 4; ++j) gp[j] = lp[row * 16 + ch * 8 + j];
    }
    // attn @ V: rebuild A tiles (bf16) from LDS, B tiles from transposed V
    #pragma unroll
    for (int t = 0; t < 2; ++t) {
      v16bf pa;
      #pragma unroll
      for (int i = 0; i < 8; ++i) {
        pa[i]     = (__bf16)pt[lh * 64 + t * 32 + hf * 16 + i];
        pa[i + 8] = (__bf16)pt[lh * 64 + t * 32 + hf * 16 + 8 + i];
      }
      #pragma unroll
      for (int dt = 0; dt < 4; ++dt) {
        const __bf16* vrow = vb + (size_t)(dt * 16 + lh) * L_ + kc + t * 32 + hf * 16;
        v16bf bv = ld_b16(vrow);
        oacc[dt] = wmma_bf16(pa, bv, oacc[dt]);
      }
    }
    __syncthreads();
  }
  // store per-head output (bf16) for the final projection
  #pragma unroll
  for (int dt = 0; dt < 4; ++dt) {
    #pragma unroll
    for (int r = 0; r < 8; ++r) {
      size_t row = (size_t)n * L_ + q0 + r + 8 * hf;
      oh[row * E_ + h * D_ + dt * 16 + lh] = (__bf16)(oacc[dt][r]);
    }
  }
}

// ---------------------------------------------------------------------------
// out = oh @ W_out^T + b_out  (4096x1024 @ 1024x1024), WMMA bf16.
__global__ __launch_bounds__(32) void mqa_proj_out(
    const __bf16* __restrict__ oh, const __bf16* __restrict__ wob,
    const float* __restrict__ bias, float* __restrict__ out) {
  int bid = blockIdx.x;
  int ct = bid & 15;            // 16 column tiles of 64
  int rt = bid >> 4;            // 256 row tiles of 16
  int lane = threadIdx.x;
  int lh = lane & 15, hf = lane >> 4;
  int r0 = rt * 16, c0 = ct * 64;

  v8f acc[4];
  #pragma unroll
  for (int i = 0; i < 4; ++i) acc[i] = vzero8();

  for (int kc = 0; kc < E_; kc += 32) {
    const __bf16* arow = oh + (size_t)(r0 + lh) * E_ + kc;
    v16bf a = ld_a16(arow, hf);
    #pragma unroll
    for (int nt = 0; nt < 4; ++nt) {
      const __bf16* brow = wob + (size_t)(c0 + nt * 16 + lh) * E_ + kc + hf * 16;
      v16bf b = ld_b16(brow);
      acc[nt] = wmma_bf16(a, b, acc[nt]);
    }
  }
  #pragma unroll
  for (int nt = 0; nt < 4; ++nt) {
    #pragma unroll
    for (int r = 0; r < 8; ++r) {
      int row = r0 + r + 8 * hf;
      int col = c0 + nt * 16 + lh;
      out[(size_t)row * E_ + col] = acc[nt][r] + bias[col];
    }
  }
}

// ---------------------------------------------------------------------------
extern "C" void kernel_launch(void* const* d_in, const int* in_sizes, int n_in,
                              void* d_out, int out_size, void* d_ws, size_t ws_size,
                              hipStream_t stream) {
  (void)in_sizes; (void)n_in; (void)out_size; (void)ws_size;
  const float* values  = (const float*)d_in[0];
  const float* keys    = (const float*)d_in[1];
  const float* queries = (const float*)d_in[2];
  const float* Wv      = (const float*)d_in[3];
  const float* Wk      = (const float*)d_in[4];
  const float* Wq      = (const float*)d_in[5];
  const float* Wout    = (const float*)d_in[6];
  const float* bout    = (const float*)d_in[7];

  float* out = (float*)d_out;                       // [N, L, E]
  float* att = out + (size_t)N_ * L_ * E_;          // [N, H, L, L]

  char* ws = (char*)d_ws;
  __bf16* qb  = (__bf16*)ws; ws += (size_t)N_ * L_ * E_ * 2;   // q proj (scaled)
  __bf16* kb  = (__bf16*)ws; ws += (size_t)N_ * L_ * D_ * 2;   // k proj
  __bf16* vbt = (__bf16*)ws; ws += (size_t)N_ * L_ * D_ * 2;   // v proj, transposed
  __bf16* wob = (__bf16*)ws; ws += (size_t)E_ * E_ * 2;        // W_out bf16
  __bf16* oh  = (__bf16*)ws; ws += (size_t)N_ * L_ * E_ * 2;   // attn@V result
  float*  rmx = (float*)ws;  ws += (size_t)N_ * H_ * L_ * 4;   // row max
  float*  rsm = (float*)ws;  ws += (size_t)N_ * H_ * L_ * 4;   // row sumexp

  mqa_convw<<<(E_ * E_ + 255) / 256, 256, 0, stream>>>(Wout, wob, E_ * E_);
  mqa_proj_kv<<<N_ * L_, 64, 0, stream>>>(keys, values, Wk, Wv, kb, vbt);
  mqa_proj_q<<<N_ * L_, 1024, 0, stream>>>(queries, Wq, qb);
  mqa_stats<<<N_ * H_ * (L_ / 16), 32, 0, stream>>>(qb, kb, rmx, rsm);
  mqa_attn<<<N_ * H_ * (L_ / 16), 32, 0, stream>>>(qb, kb, vbt, rmx, rsm, att, oh);
  mqa_proj_out<<<(N_ * L_ / 16) * (E_ / 64), 32, 0, stream>>>(oh, wob, bout, out);
}
